// GRUObservationCell_46901042872632
// MI455X (gfx1250) — matrix operand order repaired and checked by reading, not compiled, not run on hardware
//
#include <hip/hip_runtime.h>

// ---------------------------------------------------------------------------
// Types for CDNA5 WMMA (wave32, gfx1250)
// ---------------------------------------------------------------------------
typedef __bf16 bf16;
typedef __attribute__((ext_vector_type(16))) __bf16 v16bf;
typedef __attribute__((ext_vector_type(8)))  __bf16 v8bf;
typedef __attribute__((ext_vector_type(8)))  float  v8f;

static __device__ __forceinline__ unsigned short bf_bits(bf16 b) {
  union { bf16 b; unsigned short s; } u; u.b = b; return u.s;
}
// Split fp32 into bf16 hi/lo pair packed in one u32 (lo in high 16 bits).
static __device__ __forceinline__ unsigned int split_pack(float v) {
  bf16 h = (bf16)v;
  float hf = (float)h;
  bf16 l = (bf16)(v - hf);
  return (unsigned int)bf_bits(h) | ((unsigned int)bf_bits(l) << 16);
}

// ---------------------------------------------------------------------------
// 1) h -> d_out bulk copy (float4 vectorized). N*256 is divisible by 4.
// ---------------------------------------------------------------------------
__global__ void k_copy_h(const float* __restrict__ h, float* __restrict__ out, long n4) {
  long t = (long)blockIdx.x * blockDim.x + threadIdx.x;
  if (t < n4) ((float4*)out)[t] = ((const float4*)h)[t];
}

// ---------------------------------------------------------------------------
// 2) Convert a row-major fp32 weight matrix B[K, NC] into transposed bf16
//    hi/lo matrices BT[NC, K] so GEMM lanes read contiguous K segments.
// ---------------------------------------------------------------------------
__global__ void k_conv_wT(const float* __restrict__ B,
                          unsigned short* __restrict__ Th,
                          unsigned short* __restrict__ Tl, int K, int NC) {
  long t = (long)blockIdx.x * 256 + threadIdx.x;
  if (t >= (long)K * NC) return;
  int k = (int)(t / NC), c = (int)(t % NC);
  unsigned int pk = split_pack(B[t]);
  long o = (long)c * K + k;
  Th[o] = (unsigned short)(pk & 0xffffu);
  Tl[o] = (unsigned short)(pk >> 16);
}

// ---------------------------------------------------------------------------
// 3) Gather h[i_obs] and split to bf16 hi/lo ([n, 256] row-major).
// ---------------------------------------------------------------------------
__global__ void k_conv_hA(const float* __restrict__ h, const int* __restrict__ iobs,
                          unsigned short* __restrict__ Ah, unsigned short* __restrict__ Al,
                          long n) {
  long t = (long)blockIdx.x * 256 + threadIdx.x;
  if (t >= n * 256) return;
  long j = t >> 8; int c = (int)(t & 255);
  unsigned int pk = split_pack(h[(long)iobs[j] * 256 + c]);
  Ah[t] = (unsigned short)(pk & 0xffffu);
  Al[t] = (unsigned short)(pk >> 16);
}

// ---------------------------------------------------------------------------
// 4) Prep: loss terms + 4->16 per-feature GEMM + relu + mask, emitted in the
//    TF-reshape-scrambled layout:  out_row=(p*n+r)/16, out_col=((p*n+r)%16)*64+i.
//    One 256-thread block handles 16 obs rows; the 16x1024 bf16(hi,lo) tile is
//    staged in 64KB LDS and streamed out coalesced as separate hi/lo arrays.
// ---------------------------------------------------------------------------
__global__ void k_prep(const float* __restrict__ p, const float* __restrict__ X,
                       const float* __restrict__ Mo, const float* __restrict__ w,
                       const float* __restrict__ bias, const int* __restrict__ iobs,
                       long n, unsigned short* __restrict__ gih,
                       unsigned short* __restrict__ gil, float* __restrict__ lossp) {
  __shared__ unsigned int tile[16 * 1024];  // [ph][lr*64 + i], packed hi|lo
  const int tid = threadIdx.x;
  const long q = blockIdx.x;
  const int lr = tid & 15, ig = tid >> 4;
  const long r = q * 16 + lr;
  float lsum = 0.f;
  if (r < n) {
    const long idx = iobs[r];
    #pragma unroll
    for (int f = 0; f < 4; ++f) {
      const int i = ig * 4 + f;
      const float Xv  = X[r * 64 + i];
      const float mean = p[idx * 128 + i];
      const float var  = fabsf(p[idx * 128 + 64 + i]) + 1e-6f;
      const float err  = (Xv - mean) / sqrtf(var);
      const float m    = Mo[r * 64 + i];
      lsum += 0.5f * (err * err + logf(var)) * m;
      const float* wi = w + i * 64;       // w_prep[i][4][16]
      const float* bi = bias + i * 16;
      #pragma unroll
      for (int ph = 0; ph < 16; ++ph) {
        float v = Xv * wi[ph] + mean * wi[16 + ph] + var * wi[32 + ph] +
                  err * wi[48 + ph] + bi[ph];
        v = v > 0.f ? v : 0.f;            // relu, then mask
        tile[ph * 1024 + lr * 64 + i] = split_pack(v * m);
      }
    }
  } else {
    for (int f = 0; f < 4; ++f) {
      int i = ig * 4 + f;
      for (int ph = 0; ph < 16; ++ph) tile[ph * 1024 + lr * 64 + i] = 0u;
    }
  }
  // deterministic wave32 reduction of loss partials
  float wsum = lsum;
  for (int off = 16; off > 0; off >>= 1) wsum += __shfl_down(wsum, off, 32);
  if ((tid & 31) == 0) lossp[q * 8 + (tid >> 5)] = wsum;
  __syncthreads();
  for (int e = tid; e < 16384; e += 256) {
    const int ph = e >> 10, c = e & 1023;
    const int lr2 = c >> 6, i2 = c & 63;
    const long rr = q * 16 + lr2;
    if (rr >= n) continue;
    const long flat = (long)ph * n + rr;
    const long addr = (flat >> 4) * 1024 + (flat & 15) * 64 + i2;
    const unsigned int pk = tile[e];
    gih[addr] = (unsigned short)(pk & 0xffffu);
    gil[addr] = (unsigned short)(pk >> 16);
  }
}

// ---------------------------------------------------------------------------
// 5) Deterministic final loss reduction (single block, fixed order).
// ---------------------------------------------------------------------------
__global__ void k_loss_reduce(const float* __restrict__ part, int count,
                              float* __restrict__ out_loss) {
  __shared__ float s[256];
  float acc = 0.f;
  for (int i = threadIdx.x; i < count; i += 256) acc += part[i];
  s[threadIdx.x] = acc;
  __syncthreads();
  for (int w = 128; w > 0; w >>= 1) {
    if (threadIdx.x < w) s[threadIdx.x] += s[threadIdx.x + w];
    __syncthreads();
  }
  if (threadIdx.x == 0) out_loss[0] = s[0];
}

// ---------------------------------------------------------------------------
// 6/7) bf16x3 split GEMM:  C[M,768] = A[M,K] * B[K,768], fp32 accumulate.
//   A given as bf16 hi/lo row-major [M,K]; B given transposed bf16 hi/lo
//   [768,K]. Each wave computes one 16x64 slab (4 WMMA accumulators),
//   3 v_wmma_f32_16x16x32_bf16 per 32-K step per tile. Requires M % 16 == 0
//   (n_obs = 50000 satisfies this).
// ---------------------------------------------------------------------------
static __device__ __forceinline__ v16bf load_a(const bf16* __restrict__ A, int lda,
                                               long m0, int k0, int lane) {
  // ISA 16-bit A layout: lanes 0-15 -> K 0-7 & 16-23, lanes 16-31 -> K 8-15 & 24-31.
  const int half = lane >> 4;
  const bf16* row = A + (m0 + (lane & 15)) * (long)lda + k0 + half * 8;
  v8bf s0 = *(const v8bf*)(row);
  v8bf s1 = *(const v8bf*)(row + 16);
  return __builtin_shufflevector(s0, s1, 0, 1, 2, 3, 4, 5, 6, 7,
                                 8, 9, 10, 11, 12, 13, 14, 15);
}
static __device__ __forceinline__ v16bf load_b(const bf16* __restrict__ BT, int ldk,
                                               int n0, int k0, int lane) {
  // ISA 16-bit B layout: N = lane&15; lanes 0-15 -> K 0-7 & 16-23, lanes 16-31 -> K 8-15 & 24-31.
  const int half = lane >> 4;
  const bf16* col = BT + (long)(n0 + (lane & 15)) * ldk + k0 + half * 8;
  v8bf s0 = *(const v8bf*)(col);
  v8bf s1 = *(const v8bf*)(col + 16);
  return __builtin_shufflevector(s0, s1, 0, 1, 2, 3, 4, 5, 6, 7,
                                 8, 9, 10, 11, 12, 13, 14, 15);
}

__global__ void __launch_bounds__(256)
k_gemm(const unsigned short* __restrict__ Ahi_, const unsigned short* __restrict__ Alo_,
       const unsigned short* __restrict__ Bhi_, const unsigned short* __restrict__ Blo_,
       float* __restrict__ C, long M, int K) {
  const bf16* Ahi = (const bf16*)Ahi_;
  const bf16* Alo = (const bf16*)Alo_;
  const bf16* Bhi = (const bf16*)Bhi_;
  const bf16* Blo = (const bf16*)Blo_;
  const int lane = threadIdx.x;                        // 0..31
  const long waveJob = (long)blockIdx.x * 8 + threadIdx.y;
  const long mTiles = M >> 4;
  const long total = mTiles * 12;                      // 768/64 = 12 N-quads
  if (waveJob >= total) return;                        // wave-uniform: EXEC stays all-1s
  const long mt = waveJob / 12;
  const int  nq = (int)(waveJob % 12);
  const long m0 = mt << 4;
  const int  nb = nq * 64;

  v8f acc[4] = {v8f{}, v8f{}, v8f{}, v8f{}};
  for (int k0 = 0; k0 < K; k0 += 32) {
    const v16bf ah = load_a(Ahi, K, m0, k0, lane);
    const v16bf al = load_a(Alo, K, m0, k0, lane);
    #pragma unroll
    for (int t = 0; t < 4; ++t) {
      const v16bf bh = load_b(Bhi, K, nb + t * 16, k0, lane);
      const v16bf bl = load_b(Blo, K, nb + t * 16, k0, lane);
      acc[t] = __builtin_amdgcn_wmma_f32_16x16x32_bf16(false, ah, false, bh,
                                                       (short)0, acc[t], false, false);
      acc[t] = __builtin_amdgcn_wmma_f32_16x16x32_bf16(false, al, false, bh,
                                                       (short)0, acc[t], false, false);
      acc[t] = __builtin_amdgcn_wmma_f32_16x16x32_bf16(false, ah, false, bl,
                                                       (short)0, acc[t], false, false);
    }
  }
  // C/D layout: VGPR v -> M = v + (lane>=16)*8, N = lane&15.
  const int half = lane >> 4;
  const int nn = lane & 15;
  float* Cp = C + (m0 + half * 8) * 768;
  #pragma unroll
  for (int t = 0; t < 4; ++t) {
    #pragma unroll
    for (int v = 0; v < 8; ++v) Cp[(long)v * 768 + nb + t * 16 + nn] = acc[t][v];
  }
}

// ---------------------------------------------------------------------------
// 8) GRU gates + scatter into d_out at i_obs rows.
// ---------------------------------------------------------------------------
__global__ void k_gate(const float* __restrict__ mx, const float* __restrict__ mh,
                       const float* __restrict__ gb, const float* __restrict__ h,
                       const int* __restrict__ iobs, float* __restrict__ out, long n) {
  long t = (long)blockIdx.x * 256 + threadIdx.x;
  if (t >= n * 256) return;
  const long j = t >> 8;
  const int c = (int)(t & 255);
  const long idx = iobs[j];
  const float xz = mx[j * 768 + c]       + gb[c];
  const float xr = mx[j * 768 + 256 + c] + gb[256 + c];
  const float xh = mx[j * 768 + 512 + c] + gb[512 + c];
  const float rz = mh[j * 768 + c]       + gb[768 + c];
  const float rr = mh[j * 768 + 256 + c] + gb[768 + 256 + c];
  const float rh = mh[j * 768 + 512 + c] + gb[768 + 512 + c];
  const float z = 1.f / (1.f + expf(-(xz + rz)));
  const float rg = 1.f / (1.f + expf(-(xr + rr)));
  const float hh = tanhf(xh + rg * rh);
  const float hp = h[idx * 256 + c];
  out[idx * 256 + c] = z * hp + (1.f - z) * hh;
}

// ---------------------------------------------------------------------------
// Host orchestration
// ---------------------------------------------------------------------------
extern "C" void kernel_launch(void* const* d_in, const int* in_sizes, int n_in,
                              void* d_out, int out_size, void* d_ws, size_t ws_size,
                              hipStream_t stream) {
  const float* h      = (const float*)d_in[0];
  const float* p      = (const float*)d_in[1];
  const float* X      = (const float*)d_in[2];
  const float* Mo     = (const float*)d_in[3];
  const float* w_prep = (const float*)d_in[4];
  const float* b_prep = (const float*)d_in[5];
  const float* gk     = (const float*)d_in[6];
  const float* grk    = (const float*)d_in[7];
  const float* gb     = (const float*)d_in[8];
  const int*   iobs   = (const int*)d_in[9];
  const long N = (long)in_sizes[0] / 256;
  const long n = (long)in_sizes[9];   // n_obs (50000; multiple of 16)
  float* out = (float*)d_out;

  char* base = (char*)d_ws;
  size_t off = 0;
  auto take = [&](size_t bytes) -> char* {
    char* r = base + off;
    off += (bytes + 255) & ~(size_t)255;
    return r;
  };
  unsigned short* gi_hi  = (unsigned short*)take((size_t)n * 1024 * 2);
  unsigned short* gi_lo  = (unsigned short*)take((size_t)n * 1024 * 2);
  float*          mx     = (float*)take((size_t)n * 768 * 4);
  unsigned short* hA_hi  = (unsigned short*)take((size_t)n * 256 * 2);
  unsigned short* hA_lo  = (unsigned short*)take((size_t)n * 256 * 2);
  unsigned short* BT1_hi = (unsigned short*)take((size_t)1024 * 768 * 2);
  unsigned short* BT1_lo = (unsigned short*)take((size_t)1024 * 768 * 2);
  unsigned short* BT2_hi = (unsigned short*)take((size_t)256 * 768 * 2);
  unsigned short* BT2_lo = (unsigned short*)take((size_t)256 * 768 * 2);
  const long gridPrep = (n + 15) / 16;
  float* lossp = (float*)take((size_t)gridPrep * 8 * 4);
  // mh aliases the gi hi/lo region: gi (n*4096B) is dead once GEMM1 finishes,
  // and stream ordering guarantees GEMM1 completes before GEMM2 writes here.
  float* mh = (float*)base;  // needs n*3072B <= n*4096B

  // 1) h -> out
  {
    long n4 = N * 64;  // N*256/4
    k_copy_h<<<(unsigned)((n4 + 255) / 256), 256, 0, stream>>>(h, out, n4);
  }
  // 2) weights -> transposed bf16 hi/lo
  k_conv_wT<<<(unsigned)(((long)1024 * 768 + 255) / 256), 256, 0, stream>>>(
      gk, BT1_hi, BT1_lo, 1024, 768);
  k_conv_wT<<<(unsigned)(((long)256 * 768 + 255) / 256), 256, 0, stream>>>(
      grk, BT2_hi, BT2_lo, 256, 768);
  // 3) gathered h_obs -> bf16 hi/lo
  k_conv_hA<<<(unsigned)((n * 256 + 255) / 256), 256, 0, stream>>>(h, iobs, hA_hi, hA_lo, n);
  // 4) prep (gi + loss partials)
  k_prep<<<(unsigned)gridPrep, 256, 0, stream>>>(p, X, Mo, w_prep, b_prep, iobs, n,
                                                 gi_hi, gi_lo, lossp);
  // 5) loss scalar
  k_loss_reduce<<<1, 256, 0, stream>>>(lossp, (int)(gridPrep * 8), out + N * 256);
  // 6) mx = gi @ gru_kernel
  {
    long jobs = (n >> 4) * 12;
    k_gemm<<<(unsigned)((jobs + 7) / 8), dim3(32, 8, 1), 0, stream>>>(
        gi_hi, gi_lo, BT1_hi, BT1_lo, mx, n, 1024);
  }
  // 7) mh = h_obs @ gru_rec_kernel
  {
    long jobs = (n >> 4) * 12;
    k_gemm<<<(unsigned)((jobs + 7) / 8), dim3(32, 8, 1), 0, stream>>>(
        hA_hi, hA_lo, BT2_hi, BT2_lo, mh, n, 256);
  }
  // 8) gates + scatter
  k_gate<<<(unsigned)((n * 256 + 255) / 256), 256, 0, stream>>>(mx, mh, gb, h, iobs, out, n);
}
